// LlamaAttention_83288005804041
// MI455X (gfx1250) — compile-verified, matrix operand order checked by reading, not run on hardware
//
#include <hip/hip_runtime.h>
#include <cstdint>

typedef __attribute__((ext_vector_type(16))) __bf16 v16bf;
typedef __attribute__((ext_vector_type(8)))  __bf16 v8bf;
typedef __attribute__((ext_vector_type(8)))  float  v8f;

constexpr int HSZ = 4096;   // hidden size
constexpr int NH  = 32;     // heads
constexpr int HD  = 128;    // head dim
constexpr int SEQ = 4096;   // sequence length

__device__ __forceinline__ v16bf pack16(v8bf lo, v8bf hi) {
  v16bf r;
#pragma unroll
  for (int e = 0; e < 8; ++e) { r[e] = lo[e]; r[e + 8] = hi[e]; }
  return r;
}

// issue one 16-byte async global->LDS copy (ASYNCcnt-tracked)
__device__ __forceinline__ void async_copy_b128(void* lds, const void* gptr) {
  asm volatile("global_load_async_to_lds_b128 %0, %1, off"
               :: "v"((uint32_t)(uintptr_t)lds), "v"(gptr)
               : "memory");
}
__device__ __forceinline__ void wait_async0() {
  asm volatile("s_wait_asynccnt 0x0" ::: "memory");
}

// ---------------------------------------------------------------------------
// fp32 -> bf16 bulk convert (8 elements / thread)
// ---------------------------------------------------------------------------
__global__ void __launch_bounds__(256)
cvt_bf16(const float* __restrict__ in, __bf16* __restrict__ out)
{
  const size_t i = ((size_t)blockIdx.x * 256 + threadIdx.x) * 8;
  float4 a = *(const float4*)(in + i);
  float4 b = *(const float4*)(in + i + 4);
  v8bf v;
  v[0] = (__bf16)a.x; v[1] = (__bf16)a.y; v[2] = (__bf16)a.z; v[3] = (__bf16)a.w;
  v[4] = (__bf16)b.x; v[5] = (__bf16)b.y; v[6] = (__bf16)b.z; v[7] = (__bf16)b.w;
  *(v8bf*)(out + i) = v;
}

// ---------------------------------------------------------------------------
// fp32 [K][N] -> bf16 [N][K] (convert + transpose, 64x64 tiles via LDS)
// ---------------------------------------------------------------------------
__global__ void __launch_bounds__(256)
cvt_transpose_bf16(const float* __restrict__ in, __bf16* __restrict__ out)
{
  constexpr int TD = 64, STR = TD + 8;     // 144B rows: 16B aligned
  __shared__ __bf16 t[TD][STR];            // [n][k]
  const int k0 = blockIdx.x * TD;
  const int n0 = blockIdx.y * TD;
  const int tid = threadIdx.x;
  {
    const int kr = tid >> 4;               // 0..15
    const int nc = (tid & 15) * 4;         // 0..60
#pragma unroll
    for (int i = 0; i < 4; ++i) {
      const int k = kr + i * 16;
      float4 v = *(const float4*)(in + (size_t)(k0 + k) * HSZ + n0 + nc);
      t[nc + 0][k] = (__bf16)v.x;
      t[nc + 1][k] = (__bf16)v.y;
      t[nc + 2][k] = (__bf16)v.z;
      t[nc + 3][k] = (__bf16)v.w;
    }
  }
  __syncthreads();
  {
    const int nr = tid >> 2;               // 0..63
    const int kc = (tid & 3) * 16;         // 0,16,32,48
#pragma unroll
    for (int i = 0; i < 2; ++i)
      *(v8bf*)(out + (size_t)(n0 + nr) * HSZ + k0 + kc + i * 8) =
          *(const v8bf*)&t[nr][kc + i * 8];
  }
}

// ---------------------------------------------------------------------------
// GEMM: C[M,N] = A[M,K] * Bt[N,K]^T ; bf16 operands, f32 accumulate/output.
// 128x128x64 block tile, 8 waves (wave = 2x4 WMMA tiles), double-buffered
// LDS filled by async global->LDS copies overlapping the WMMA pipe.
// ---------------------------------------------------------------------------
template<int M, int N, int K>
__global__ void __launch_bounds__(256)
gemm_bf16_nt(const __bf16* __restrict__ A,    // [M][K]
             const __bf16* __restrict__ Bt,   // [N][K]
             float* __restrict__ C)           // [M][N] fp32
{
  constexpr int BM = 128, BN = 128, BK = 64, LSTR = BK + 8;   // 144B rows
  __shared__ __bf16 As[2][BM][LSTR];
  __shared__ __bf16 Bs[2][BN][LSTR];

  const int tid  = threadIdx.x;
  const int lane = tid & 31;
  const int wave = tid >> 5;
  const int l16  = lane & 15;
  const int half = lane >> 4;
  const int wrow = (wave & 3) * 32;
  const int wcol = (wave >> 2) * 64;
  const int bm = blockIdx.x * BM;
  const int bn = blockIdx.y * BN;

  const int srow = tid >> 1;              // 0..127
  const int scol = (tid & 1) * 32;        // 0 or 32
  const __bf16* ga = A  + (size_t)(bm + srow) * K + scol;
  const __bf16* gb = Bt + (size_t)(bn + srow) * K + scol;

  auto stage = [&](int buf, int k0) {
#pragma unroll
    for (int i = 0; i < 4; ++i) {
      async_copy_b128(&As[buf][srow][scol + i * 8], ga + k0 + i * 8);
      async_copy_b128(&Bs[buf][srow][scol + i * 8], gb + k0 + i * 8);
    }
  };

  v8f acc[2][4] = {};
  stage(0, 0);
  wait_async0();
  __syncthreads();

  constexpr int NIT = K / BK;
  for (int it = 0; it < NIT; ++it) {
    const int cur = it & 1;
    if (it + 1 < NIT) stage(cur ^ 1, (it + 1) * BK);
    if (it + 2 < NIT) {                    // GL2 prefetch 2 tiles ahead
      __builtin_prefetch(ga + (it + 2) * BK, 0, 1);
      __builtin_prefetch(gb + (it + 2) * BK, 0, 1);
    }

#pragma unroll
    for (int ks = 0; ks < 2; ++ks) {
      v16bf af[2], bfr[4];
#pragma unroll
      for (int mt = 0; mt < 2; ++mt) {
        const __bf16* p = &As[cur][wrow + mt * 16 + l16][ks * 32 + half * 8];
        af[mt] = pack16(*(const v8bf*)p, *(const v8bf*)(p + 16));
      }
#pragma unroll
      for (int nt = 0; nt < 4; ++nt) {
        const __bf16* p = &Bs[cur][wcol + nt * 16 + l16][ks * 32 + half * 8];
        bfr[nt] = pack16(*(const v8bf*)p, *(const v8bf*)(p + 16));
      }
#pragma unroll
      for (int mt = 0; mt < 2; ++mt)
#pragma unroll
        for (int nt = 0; nt < 4; ++nt)
          acc[mt][nt] = __builtin_amdgcn_wmma_f32_16x16x32_bf16(
              false, af[mt], false, bfr[nt], (short)0, acc[mt][nt], false, false);
    }
    wait_async0();
    __syncthreads();
  }

#pragma unroll
  for (int mt = 0; mt < 2; ++mt)
#pragma unroll
    for (int nt = 0; nt < 4; ++nt) {
      float* cp = C + (size_t)(bm + wrow + mt * 16 + half * 8) * N
                    + (bn + wcol + nt * 16 + l16);
#pragma unroll
      for (int r = 0; r < 8; ++r) cp[(size_t)r * N] = acc[mt][nt][r];
    }
}

// ---------------------------------------------------------------------------
// RoPE on Q,K + repack to [NH, T, HD]; emits fp32 keys/values outputs and
// bf16 q/k/v for the attention kernel.
// ---------------------------------------------------------------------------
__global__ void __launch_bounds__(256)
rope_pack(const float* __restrict__ Qf, const float* __restrict__ Kf,
          const float* __restrict__ Vf, const int* __restrict__ pos,
          float* __restrict__ keys_out, float* __restrict__ vals_out,
          __bf16* __restrict__ qb, __bf16* __restrict__ kb,
          __bf16* __restrict__ vb)
{
  const int idx = blockIdx.x * blockDim.x + threadIdx.x;
  const int j = idx & 63;
  const int h = (idx >> 6) & 31;
  const int t = idx >> 11;
  if (t >= SEQ) return;

  const float inv = __powf(10000.0f, -(float)(2 * j) * (1.0f / (float)HD));
  const float ang = (float)pos[t] * inv;
  float sn, cs;
  __sincosf(ang, &sn, &cs);

  const size_t s1 = (size_t)t * HSZ + (size_t)h * HD + j;
  const size_t s2 = s1 + 64;
  const size_t d1 = ((size_t)h * SEQ + t) * HD + j;
  const size_t d2 = d1 + 64;

  const float q1 = Qf[s1], q2 = Qf[s2];
  const float k1 = Kf[s1], k2 = Kf[s2];
  const float qo1 = q1 * cs - q2 * sn, qo2 = q2 * cs + q1 * sn;
  const float ko1 = k1 * cs - k2 * sn, ko2 = k2 * cs + k1 * sn;

  qb[d1] = (__bf16)qo1; qb[d2] = (__bf16)qo2;
  kb[d1] = (__bf16)ko1; kb[d2] = (__bf16)ko2;
  keys_out[d1] = ko1;   keys_out[d2] = ko2;

  const float v1 = Vf[s1], v2 = Vf[s2];
  vb[d1] = (__bf16)v1;  vb[d2] = (__bf16)v2;
  vals_out[d1] = v1;    vals_out[d2] = v2;
}

// ---------------------------------------------------------------------------
// Flash attention (prefill): block = 1 head x 64 query rows, 4 waves,
// each wave owns 16 rows. Streams 64-key tiles with online softmax.
// K tile staged by async global->LDS copies; V staged transposed.
// Fragment streams are software-pipelined (cur/next ping-pong) so the
// ds_load_b128 latency hides under the previous WMMA.
// ---------------------------------------------------------------------------
__global__ void __launch_bounds__(128)
flash_attn_bf16(const __bf16* __restrict__ qb, const __bf16* __restrict__ kb,
                const __bf16* __restrict__ vb, const float* __restrict__ mask,
                __bf16* __restrict__ attn_out)
{
  constexpr int BQ = 64, BC = 64;
  __shared__ __bf16 Ks[BC][HD + 16];     // [key][d]  (B-frag layout for QK^T)
  __shared__ __bf16 Vs[HD][BC + 16];     // [d][key]  (B-frag layout for P*V)
  __shared__ __bf16 Ps[4][16][BC];       // per-wave P staging

  const int tid  = threadIdx.x;
  const int lane = tid & 31;
  const int wave = tid >> 5;
  const int l16  = lane & 15;
  const int half = lane >> 4;
  const int head = blockIdx.y;
  const int q0   = blockIdx.x * BQ;

  const __bf16* qh = qb + (size_t)head * SEQ * HD;
  const __bf16* kh = kb + (size_t)head * SEQ * HD;
  const __bf16* vh = vb + (size_t)head * SEQ * HD;

  v16bf qf[4];
  {
    const int qrow = q0 + wave * 16 + l16;
#pragma unroll
    for (int ks = 0; ks < 4; ++ks) {
      const __bf16* p = qh + (size_t)qrow * HD + ks * 32 + half * 8;
      qf[ks] = pack16(*(const v8bf*)p, *(const v8bf*)(p + 16));
    }
  }

  v8f o[8] = {};
  float mrow[8], lrow[8];
#pragma unroll
  for (int r = 0; r < 8; ++r) { mrow[r] = -__builtin_inff(); lrow[r] = 0.f; }
  const float scale = 0.0883883476483184f;   // 1/sqrt(128)
  const int rbase = q0 + wave * 16 + half * 8;

  const int srow  = tid >> 1;           // 0..63
  const int scolb = (tid & 1) * 64;     // 0 or 64

  auto kfrag = [&](int ks, int nt) -> v16bf {
    const __bf16* p = &Ks[nt * 16 + l16][ks * 32 + half * 8];
    return pack16(*(const v8bf*)p, *(const v8bf*)(p + 16));
  };
  auto vfrag = [&](int ks, int ct) -> v16bf {
    const __bf16* p = &Vs[ct * 16 + l16][ks * 32 + half * 8];
    return pack16(*(const v8bf*)p, *(const v8bf*)(p + 16));
  };

  for (int c0 = 0; c0 < SEQ; c0 += BC) {
    {
      const __bf16* ksrc = kh + (size_t)(c0 + srow) * HD + scolb;
      const __bf16* vsrc = vh + (size_t)(c0 + srow) * HD + scolb;
#pragma unroll
      for (int i = 0; i < 8; ++i)
        async_copy_b128(&Ks[srow][scolb + i * 8], ksrc + i * 8);
#pragma unroll
      for (int i = 0; i < 8; ++i) {
        v8bf vv = *(const v8bf*)(vsrc + i * 8);
#pragma unroll
        for (int j = 0; j < 8; ++j) Vs[scolb + i * 8 + j][srow] = vv[j];
      }
      if (c0 + BC < SEQ) {
        __builtin_prefetch(ksrc + BC * HD, 0, 1);
        __builtin_prefetch(vsrc + BC * HD, 0, 1);
      }
    }
    wait_async0();
    __syncthreads();

    // S = Q * K^T  (16 x 64 per wave), pipelined fragment stream
    v8f s[4] = {};
    {
      v16bf bcur = kfrag(0, 0);
#pragma unroll
      for (int idx = 0; idx < 16; ++idx) {
        const int ks = idx >> 2, nt = idx & 3;
        v16bf bnxt = bcur;
        if (idx + 1 < 16) bnxt = kfrag((idx + 1) >> 2, (idx + 1) & 3);
        s[nt] = __builtin_amdgcn_wmma_f32_16x16x32_bf16(
            false, qf[ks], false, bcur, (short)0, s[nt], false, false);
        bcur = bnxt;
      }
    }

    // scale + mask + online softmax (row = 16 lanes within half-wave)
#pragma unroll
    for (int r = 0; r < 8; ++r) {
      const size_t mbase = (size_t)(rbase + r) * SEQ + c0;
      float rm = -__builtin_inff();
#pragma unroll
      for (int nt = 0; nt < 4; ++nt) {
        float v = s[nt][r] * scale + mask[mbase + nt * 16 + l16];
        s[nt][r] = v;
        rm = fmaxf(rm, v);
      }
#pragma unroll
      for (int off = 8; off >= 1; off >>= 1)
        rm = fmaxf(rm, __shfl_xor(rm, off, 16));
      const float mnew  = fmaxf(mrow[r], rm);
      const float alpha = __expf(mrow[r] - mnew);
      mrow[r] = mnew;
      float rs = 0.f;
#pragma unroll
      for (int nt = 0; nt < 4; ++nt) {
        float p = __expf(s[nt][r] - mnew);
        s[nt][r] = p;
        rs += p;
      }
#pragma unroll
      for (int off = 8; off >= 1; off >>= 1)
        rs += __shfl_xor(rs, off, 16);
      lrow[r] = lrow[r] * alpha + rs;
#pragma unroll
      for (int ct = 0; ct < 8; ++ct) o[ct][r] *= alpha;
    }

    // P: D-layout regs -> bf16 A-layout via wave-private LDS
#pragma unroll
    for (int nt = 0; nt < 4; ++nt)
#pragma unroll
      for (int r = 0; r < 8; ++r)
        Ps[wave][half * 8 + r][nt * 16 + l16] = (__bf16)s[nt][r];

    // O += P * V, pipelined fragment stream
    {
      v16bf pa[2];
#pragma unroll
      for (int ks = 0; ks < 2; ++ks) {
        const __bf16* pp = &Ps[wave][l16][ks * 32 + half * 8];
        pa[ks] = pack16(*(const v8bf*)pp, *(const v8bf*)(pp + 16));
      }
      v16bf vcur = vfrag(0, 0);
#pragma unroll
      for (int idx = 0; idx < 16; ++idx) {
        const int ks = idx >> 3, ct = idx & 7;
        v16bf vnxt = vcur;
        if (idx + 1 < 16) vnxt = vfrag((idx + 1) >> 3, (idx + 1) & 7);
        o[ct] = __builtin_amdgcn_wmma_f32_16x16x32_bf16(
            false, pa[ks], false, vcur, (short)0, o[ct], false, false);
        vcur = vnxt;
      }
    }
    __syncthreads();
  }

  // epilogue: normalize, write bf16 [t, H] for output projection
#pragma unroll
  for (int ct = 0; ct < 8; ++ct)
#pragma unroll
    for (int r = 0; r < 8; ++r) {
      const int t = rbase + r;
      const int d = ct * 16 + l16;
      attn_out[(size_t)t * HSZ + (size_t)head * HD + d] =
          (__bf16)(o[ct][r] / lrow[r]);
    }
}

// ---------------------------------------------------------------------------
extern "C" void kernel_launch(void* const* d_in, const int* in_sizes, int n_in,
                              void* d_out, int out_size, void* d_ws, size_t ws_size,
                              hipStream_t stream) {
  (void)in_sizes; (void)n_in; (void)out_size; (void)ws_size;
  const float* hs   = (const float*)d_in[0];
  const float* mask = (const float*)d_in[1];
  const int*   pos  = (const int*)d_in[2];
  const float* Wq   = (const float*)d_in[3];
  const float* Wk   = (const float*)d_in[4];
  const float* Wv   = (const float*)d_in[5];
  const float* Wo   = (const float*)d_in[6];

  const size_t TH = (size_t)SEQ * HSZ;
  float* result = (float*)d_out;          // [T, H]
  float* keys   = result + TH;            // [NH, T, HD] fp32
  float* vals   = keys + TH;              // [NH, T, HD] fp32

  // workspace layout
  float*  Qf  = (float*)d_ws;             // [T, H] fp32 pre-rope Q
  float*  Kf  = Qf + TH;
  float*  Vf  = Kf + TH;
  __bf16* hsb = (__bf16*)(Vf + TH);       // [T, H] bf16 hidden states
  __bf16* wqt = hsb + TH;                 // [N][K] bf16 transposed weights
  __bf16* wkt = wqt + TH;
  __bf16* wvt = wkt + TH;
  __bf16* wot = wvt + TH;
  __bf16* qb  = wot + TH;                 // [NH, T, HD] bf16 post-rope
  __bf16* kb  = qb + TH;
  __bf16* vb  = kb + TH;
  __bf16* attnb = (__bf16*)Qf;            // alias: Qf dead after rope_pack

  // one-time precision conversion / weight transpose
  cvt_bf16<<<(int)(TH / (8 * 256)), 256, 0, stream>>>(hs, hsb);
  dim3 gt(HSZ / 64, HSZ / 64);
  cvt_transpose_bf16<<<gt, 256, 0, stream>>>(Wq, wqt);
  cvt_transpose_bf16<<<gt, 256, 0, stream>>>(Wk, wkt);
  cvt_transpose_bf16<<<gt, 256, 0, stream>>>(Wv, wvt);
  cvt_transpose_bf16<<<gt, 256, 0, stream>>>(Wo, wot);

  dim3 gg(SEQ / 128, HSZ / 128);
  gemm_bf16_nt<SEQ, HSZ, HSZ><<<gg, 256, 0, stream>>>(hsb, wqt, Qf);
  gemm_bf16_nt<SEQ, HSZ, HSZ><<<gg, 256, 0, stream>>>(hsb, wkt, Kf);
  gemm_bf16_nt<SEQ, HSZ, HSZ><<<gg, 256, 0, stream>>>(hsb, wvt, Vf);

  rope_pack<<<(SEQ * NH * (HD / 2)) / 256, 256, 0, stream>>>(
      Qf, Kf, Vf, pos, keys, vals, qb, kb, vb);

  flash_attn_bf16<<<dim3(SEQ / 64, NH), 128, 0, stream>>>(
      qb, kb, vb, mask, attnb);

  gemm_bf16_nt<SEQ, HSZ, HSZ><<<gg, 256, 0, stream>>>(attnb, wot, result);
}
